// HierarchicalAttention_53996328845617
// MI455X (gfx1250) — matrix operand order repaired and checked by reading, not compiled, not run on hardware
//
#include <hip/hip_runtime.h>
#include <stdint.h>

typedef __attribute__((ext_vector_type(16))) _Float16 v16h;
typedef __attribute__((ext_vector_type(8)))  float    v8f;
typedef __attribute__((ext_vector_type(4)))  int      v4i;

union V16H {
  v16h h;
  _Float16 f[16];
  unsigned u[8];
};

#define NHEADS 16
#define SEQ    8192
#define DMODEL 64
#define WWIN   128
#define NEGINF (-1e30f)

// ---------------------------------------------------------------- staging ----
#if __has_builtin(__builtin_amdgcn_global_load_async_to_lds_b128)
#define HAVE_ASYNC_LDS 1
#else
#define HAVE_ASYNC_LDS 0
#endif

typedef __attribute__((address_space(1))) v4i g_v4i;   // global int4
typedef __attribute__((address_space(3))) v4i l_v4i;   // LDS int4

// Copy 16 contiguous bytes global -> LDS for this lane. Async (DMA, ASYNCcnt)
// when the gfx1250 builtin is available, otherwise a b128 VGPR round trip.
__device__ __forceinline__ void stage16(_Float16* ldst, const _Float16* gsrc) {
#if HAVE_ASYNC_LDS
  __builtin_amdgcn_global_load_async_to_lds_b128(
      (g_v4i*)(uintptr_t)gsrc,
      (l_v4i*)(uintptr_t)ldst,   // generic LDS addr: low 32 bits = LDS offset
      0, 0);
#else
  *(uint4*)ldst = *(const uint4*)gsrc;
#endif
}
__device__ __forceinline__ void stage_fence() {
#if HAVE_ASYNC_LDS
  asm volatile("s_wait_asynccnt 0x0" ::: "memory");
#endif
}

// ------------------------------------------------------- half-wave reduces ----
#if __has_builtin(__builtin_amdgcn_ds_swizzle)
#define SWZ(x, imm) __int_as_float(__builtin_amdgcn_ds_swizzle(__float_as_int(x), (imm)))
__device__ __forceinline__ float half_reduce_max(float x) {
  x = fmaxf(x, SWZ(x, 0x041f));   // xor 1
  x = fmaxf(x, SWZ(x, 0x081f));   // xor 2
  x = fmaxf(x, SWZ(x, 0x101f));   // xor 4
  x = fmaxf(x, SWZ(x, 0x201f));   // xor 8
  return x;
}
__device__ __forceinline__ float half_reduce_sum(float x) {
  x += SWZ(x, 0x041f);
  x += SWZ(x, 0x081f);
  x += SWZ(x, 0x101f);
  x += SWZ(x, 0x201f);
  return x;
}
#else
__device__ __forceinline__ float half_reduce_max(float x) {
  x = fmaxf(x, __shfl_xor(x, 1, 32));
  x = fmaxf(x, __shfl_xor(x, 2, 32));
  x = fmaxf(x, __shfl_xor(x, 4, 32));
  x = fmaxf(x, __shfl_xor(x, 8, 32));
  return x;
}
__device__ __forceinline__ float half_reduce_sum(float x) {
  x += __shfl_xor(x, 1, 32);
  x += __shfl_xor(x, 2, 32);
  x += __shfl_xor(x, 4, 32);
  x += __shfl_xor(x, 8, 32);
  return x;
}
#endif

// ------------------------------------------------------------ attend tile ----
// One 32-key (or 32-chunk) tile vs this wave's 16-query WMMA tile, online
// softmax merged into (mrow, lrow, acc).
//   lds_k : 32 x 64 f16, key-major  (B fragments for Q*K^T)
//   lds_vt: 64 x 32 f16, dim-major  (B fragments for P*V)
//   lds_p : wave-private 16 x 32 f16 (C-layout f32 -> A-layout f16)
template<bool LVL0>
__device__ __forceinline__ void attend_tile(
    const _Float16* lds_k, const _Float16* lds_vt, _Float16* lds_p,
    const V16H& aq0, const V16H& aq1,
    int qrow0, int keybase, int S, int C, float gamma, float scale,
    float* mrow, float* lrow, v8f* acc, int lane)
{
  const int lh = lane >> 4;
  const int ln = lane & 15;

  // scores: S(16x32) = Q(16x64) @ K^T(64x32)
  v8f s[2];
  #pragma unroll
  for (int st = 0; st < 2; ++st) {
    V16H b0, b1;
    const unsigned* kp = (const unsigned*)(lds_k + (size_t)(st*16 + ln)*DMODEL + lh*16);
    #pragma unroll
    for (int i = 0; i < 8; ++i) { b0.u[i] = kp[i]; b1.u[i] = kp[i + 16]; }
    v8f z = {};
    v8f sv = __builtin_amdgcn_wmma_f32_16x16x32_f16(false, aq0.h, false, b0.h,
                                                    (short)0, z, false, false);
    sv = __builtin_amdgcn_wmma_f32_16x16x32_f16(false, aq1.h, false, b1.h,
                                                (short)0, sv, false, false);
    s[st] = sv;
  }

  // ALiBi + visibility + online softmax (C layout: row = j + 8*lh, col = ln)
  #pragma unroll
  for (int j = 0; j < 8; ++j) {
    const int i = qrow0 + j + 8*lh;
    float sc[2];
    bool  vb[2];
    float bm = NEGINF;
    #pragma unroll
    for (int st = 0; st < 2; ++st) {
      const int kidx = keybase + st*16 + ln;
      float dist;
      bool vis;
      if (LVL0) {
        dist = (float)(i - kidx);
        vis = (kidx >= 0) && ((i - kidx) >= 0) && ((i - kidx) < WWIN);
      } else {
        const int jl = (kidx + 1)*S - 1;
        dist = (float)(i - jl);
        vis = (kidx < C) && ((i - jl) >= WWIN);
      }
      float sv = vis ? (s[st][j]*scale - gamma*dist) : NEGINF;
      sc[st] = sv;
      vb[st] = vis;
      bm = fmaxf(bm, sv);
    }
    const float rm = half_reduce_max(bm);
    const float mn = fmaxf(mrow[j], rm);
    const float alpha = __expf(mrow[j] - mn);
    mrow[j] = mn;
    float rs = 0.f;
    #pragma unroll
    for (int st = 0; st < 2; ++st) {
      const float p = vb[st] ? __expf(sc[st] - mn) : 0.f;
      rs += p;
      lds_p[(size_t)(j + 8*lh)*32 + st*16 + ln] = (_Float16)p;
    }
    lrow[j] = lrow[j]*alpha + half_reduce_sum(rs);
    #pragma unroll
    for (int t = 0; t < 4; ++t) acc[t][j] *= alpha;
  }

  // DS ops from one wave are in-order; fence compiler + wait for stores.
  asm volatile("s_wait_dscnt 0" ::: "memory");

  // P (16x32, A layout) @ V (32x64) -> 4 output dim tiles
  V16H ap;
  #pragma unroll
  for (int i = 0; i < 8; ++i) {
    ap.f[i]     = lds_p[(size_t)ln*32 + lh*16 + i];
    ap.f[i + 8] = lds_p[(size_t)ln*32 + lh*16 + 8 + i];
  }
  #pragma unroll
  for (int t = 0; t < 4; ++t) {
    V16H bv;
    const unsigned* vp = (const unsigned*)(lds_vt + (size_t)(t*16 + ln)*32 + lh*16);
    #pragma unroll
    for (int i = 0; i < 8; ++i) bv.u[i] = vp[i];
    acc[t] = __builtin_amdgcn_wmma_f32_16x16x32_f16(false, ap.h, false, bv.h,
                                                    (short)0, acc[t], false, false);
  }
}

// ------------------------------------------------------------- main kernel ----
// FULLPIPE: level-0 K/V were pre-converted/pre-transposed to f16 in the
// workspace, so every tile stage is two contiguous 4 KB (async) copies.
template<bool FULLPIPE>
__global__ __launch_bounds__(256) void hattn_fused(
    const float* __restrict__ q, const float* __restrict__ k,
    const float* __restrict__ v, const float* __restrict__ gam,
    const _Float16* __restrict__ k1, const _Float16* __restrict__ v1t,
    const _Float16* __restrict__ k2, const _Float16* __restrict__ v2t,
    const _Float16* __restrict__ k3, const _Float16* __restrict__ v3t,
    const _Float16* __restrict__ kh,   // [16][8192][64]  (FULLPIPE only)
    const _Float16* __restrict__ vt0,  // [16][256][64][32] (FULLPIPE only)
    float* __restrict__ out)
{
  __shared__ __align__(16) _Float16 lds_k[32*DMODEL];
  __shared__ __align__(16) _Float16 lds_vt[DMODEL*32];
  __shared__ __align__(16) _Float16 lds_p[8][16*32];

  const int ib   = blockIdx.x;
  const int h    = blockIdx.y;
  const int tid  = threadIdx.x;
  const int wave = tid >> 5;
  const int lane = tid & 31;
  const int lh   = lane >> 4;
  const int ln   = lane & 15;

  const float scale = 0.125f;
  const float g0 = gam[0], g1v = gam[1], g2v = gam[2], g3v = gam[3];

  const int qrow0 = ib*128 + wave*16;

  // Q A-fragments (fp32 -> fp16), registers for the whole kernel.
  V16H aq0, aq1;
  {
    const float* qr = q + ((size_t)h*SEQ + (qrow0 + ln))*DMODEL + lh*16;
    #pragma unroll
    for (int i = 0; i < 8; ++i) {
      aq0.f[i]     = (_Float16)qr[i];
      aq0.f[i + 8] = (_Float16)qr[8 + i];
      aq1.f[i]     = (_Float16)qr[32 + i];
      aq1.f[i + 8] = (_Float16)qr[40 + i];
    }
  }

  float mrow[8], lrow[8];
  v8f acc[4];
  {
    v8f zero = {};
    #pragma unroll
    for (int j = 0; j < 8; ++j) { mrow[j] = NEGINF; lrow[j] = 0.f; }
    #pragma unroll
    for (int t = 0; t < 4; ++t) acc[t] = zero;
  }

  const int r  = tid >> 3;          // manual staging row (0..31)
  const int d0 = (tid & 7) * 8;     // manual staging dim group

  // ---- Level 0: tiles of 32 keys covering [(ib-1)W, (ib+1)W); tiles that are
  //      entirely j<0 (block 0, t8<4) are skipped.
  const int t8s = (ib == 0) ? 4 : 0;
  for (int t8 = t8s; t8 < 8; ++t8) {
    const int j0 = (ib - 1)*WWIN + t8*32;   // multiple of 32, >= 0 here
    __syncthreads();
    if (FULLPIPE) {
      stage16(&lds_k[tid*8],  kh  + ((size_t)h*SEQ + j0)*DMODEL + tid*8);
      stage16(&lds_vt[tid*8], vt0 + ((size_t)h*256 + (j0 >> 5))*2048 + tid*8);
      stage_fence();
    } else {
      const float* kr = k + ((size_t)h*SEQ + (j0 + r))*DMODEL + d0;
      const float* vr = v + ((size_t)h*SEQ + (j0 + r))*DMODEL + d0;
      #pragma unroll
      for (int i = 0; i < 8; ++i) {
        lds_k[(size_t)r*DMODEL + d0 + i] = (_Float16)kr[i];
        lds_vt[(size_t)(d0 + i)*32 + r]  = (_Float16)vr[i];
      }
      __builtin_prefetch(kr + 32*DMODEL, 0, 1);
      __builtin_prefetch(vr + 32*DMODEL, 0, 1);
    }
    __syncthreads();
    attend_tile<true>(lds_k, lds_vt, lds_p[wave], aq0, aq1,
                      qrow0, j0, 1, SEQ, g0, scale, mrow, lrow, acc, lane);
  }

  // ---- Hierarchical levels 1..3: pooled f16 K [h][Cpad][64] and V
  //      pre-transposed per 32-chunk tile [h][Cpad/32][64][32].
  #pragma unroll
  for (int lvl = 0; lvl < 3; ++lvl) {
    const _Float16* kl  = (lvl == 0) ? k1  : (lvl == 1) ? k2  : k3;
    const _Float16* vlt = (lvl == 0) ? v1t : (lvl == 1) ? v2t : v3t;
    const int S    = (lvl == 0) ? 8    : (lvl == 1) ? 64  : 512;
    const int C    = (lvl == 0) ? 1024 : (lvl == 1) ? 128 : 16;
    const int Cpad = (lvl == 0) ? 1024 : (lvl == 1) ? 128 : 32;
    const float gg = (lvl == 0) ? g1v  : (lvl == 1) ? g2v : g3v;

    int nvis = (ib*128) / S;              // chunks visible to some query here
    if (nvis > C) nvis = C;
    const int ntiles = (nvis + 31) >> 5;  // <= Cpad/32 by construction
    for (int t32 = 0; t32 < ntiles; ++t32) {
      const int c0 = t32*32;
      __syncthreads();
      stage16(&lds_k[tid*8],  kl  + ((size_t)h*Cpad + c0)*DMODEL + tid*8);
      stage16(&lds_vt[tid*8], vlt + ((size_t)h*(Cpad >> 5) + t32)*2048 + tid*8);
      stage_fence();
      __syncthreads();
      attend_tile<false>(lds_k, lds_vt, lds_p[wave], aq0, aq1,
                         qrow0, c0, S, C, gg, scale, mrow, lrow, acc, lane);
    }
  }

  // ---- epilogue: O = acc / max(l, 1e-8)
  #pragma unroll
  for (int t = 0; t < 4; ++t) {
    #pragma unroll
    for (int j = 0; j < 8; ++j) {
      const int i = qrow0 + j + 8*lh;
      const float lv = fmaxf(lrow[j], 1e-8f);
      out[((size_t)h*SEQ + i)*DMODEL + t*16 + ln] = acc[t][j] / lv;
    }
  }
}

// ------------------------------------------------------------- pre-passes ----
// Mean-pool S raw tokens/chunk into f16 K [h][Cpad][64]; zero-pad c >= C.
__global__ __launch_bounds__(256) void pool_k_mean(
    const float* __restrict__ src, _Float16* __restrict__ dst,
    int S, int C, int Cpad)
{
  const int idx = blockIdx.x*256 + threadIdx.x;
  const int total = NHEADS*Cpad*DMODEL;
  if (idx >= total) return;
  const int dim = idx & (DMODEL - 1);
  const int c   = (idx >> 6) % Cpad;
  const int hh  = idx / (Cpad*DMODEL);
  float a = 0.f;
  if (c < C) {
    const float* s = src + ((size_t)hh*SEQ + (size_t)c*S)*DMODEL + dim;
    for (int t = 0; t < S; ++t) a += s[(size_t)t*DMODEL];
    a /= (float)S;
  }
  dst[idx] = (_Float16)a;
}

// Mean-pool into f16 V, transposed per 32-chunk tile: [h][Cpad/32][64][32].
__global__ __launch_bounds__(256) void pool_v_trans(
    const float* __restrict__ src, _Float16* __restrict__ dst,
    int S, int C, int Cpad)
{
  const int idx = blockIdx.x*256 + threadIdx.x;
  const int total = NHEADS*Cpad*DMODEL;
  if (idx >= total) return;
  const int kk  = idx & 31;
  const int dim = (idx >> 5) & 63;
  const int t   = (idx >> 11) % (Cpad >> 5);
  const int hh  = idx / (Cpad*DMODEL);
  const int c   = t*32 + kk;
  float a = 0.f;
  if (c < C) {
    const float* s = src + ((size_t)hh*SEQ + (size_t)c*S)*DMODEL + dim;
    for (int tt = 0; tt < S; ++tt) a += s[(size_t)tt*DMODEL];
    a /= (float)S;
  }
  dst[idx] = (_Float16)a;
}

// fp32 K -> f16 K, same layout.
__global__ __launch_bounds__(256) void conv_half(
    const float* __restrict__ src, _Float16* __restrict__ dst)
{
  const int idx = blockIdx.x*256 + threadIdx.x;
  if (idx >= NHEADS*SEQ*DMODEL) return;
  dst[idx] = (_Float16)src[idx];
}

// fp32 V -> f16, transposed per 32-token tile: [h][256][64][32].
__global__ __launch_bounds__(256) void trans_v_raw(
    const float* __restrict__ src, _Float16* __restrict__ dst)
{
  const int idx = blockIdx.x*256 + threadIdx.x;
  if (idx >= NHEADS*SEQ*DMODEL) return;
  const int kk  = idx & 31;
  const int dim = (idx >> 5) & 63;
  const int m   = (idx >> 11) & 255;
  const int hh  = idx >> 19;
  dst[idx] = (_Float16)src[((size_t)hh*SEQ + m*32 + kk)*DMODEL + dim];
}

// ---------------------------------------------------------------- launcher ----
extern "C" void kernel_launch(void* const* d_in, const int* in_sizes, int n_in,
                              void* d_out, int out_size, void* d_ws, size_t ws_size,
                              hipStream_t stream)
{
  (void)in_sizes; (void)n_in; (void)out_size;

  const float* q   = (const float*)d_in[0];
  const float* k   = (const float*)d_in[1];
  const float* v   = (const float*)d_in[2];
  const float* gam = (const float*)d_in[3];

  // Workspace layout (f16 elements):
  //   k1 [16][1024][64], v1t [16][32][64][32],
  //   k2 [16][128][64],  v2t [16][4][64][32],
  //   k3 [16][32][64],   v3t [16][1][64][32],
  //   (optional) kh [16][8192][64], vt0 [16][256][64][32]
  const size_t N1 = (size_t)NHEADS*1024*DMODEL;
  const size_t N2 = (size_t)NHEADS*128*DMODEL;
  const size_t N3 = (size_t)NHEADS*32*DMODEL;
  const size_t NR = (size_t)NHEADS*SEQ*DMODEL;

  _Float16* ws  = (_Float16*)d_ws;
  _Float16* k1  = ws;
  _Float16* v1t = k1  + N1;
  _Float16* k2  = v1t + N1;
  _Float16* v2t = k2  + N2;
  _Float16* k3  = v2t + N2;
  _Float16* v3t = k3  + N3;
  _Float16* kh  = v3t + N3;
  _Float16* vt0 = kh  + NR;

  const size_t need_full = (2*(N1 + N2 + N3) + 2*NR) * sizeof(_Float16);
  const bool full = ws_size >= need_full;

  pool_k_mean <<<(int)((N1 + 255)/256), 256, 0, stream>>>(k, k1, 8, 1024, 1024);
  pool_v_trans<<<(int)((N1 + 255)/256), 256, 0, stream>>>(v, v1t, 8, 1024, 1024);
  pool_k_mean <<<(int)((N2 + 255)/256), 256, 0, stream>>>(k, k2, 64, 128, 128);
  pool_v_trans<<<(int)((N2 + 255)/256), 256, 0, stream>>>(v, v2t, 64, 128, 128);
  pool_k_mean <<<(int)((N3 + 255)/256), 256, 0, stream>>>(k, k3, 512, 16, 32);
  pool_v_trans<<<(int)((N3 + 255)/256), 256, 0, stream>>>(v, v3t, 512, 16, 32);

  if (full) {
    conv_half <<<(int)((NR + 255)/256), 256, 0, stream>>>(k, kh);
    trans_v_raw<<<(int)((NR + 255)/256), 256, 0, stream>>>(v, vt0);
    hattn_fused<true><<<dim3(SEQ/128, NHEADS), 256, 0, stream>>>(
        q, k, v, gam, k1, v1t, k2, v2t, k3, v3t, kh, vt0, (float*)d_out);
  } else {
    hattn_fused<false><<<dim3(SEQ/128, NHEADS), 256, 0, stream>>>(
        q, k, v, gam, k1, v1t, k2, v2t, k3, v3t, nullptr, nullptr, (float*)d_out);
  }
}
